// NeighbourSupport_58789512347737
// MI455X (gfx1250) — matrix operand order repaired and checked by reading, not compiled, not run on hardware
//
#include <hip/hip_runtime.h>
#include <hip/hip_bf16.h>
#include <stdint.h>

// ---- CDNA5 (gfx1250) types for WMMA ----
typedef __attribute__((ext_vector_type(16))) __bf16 v16bf;
typedef __attribute__((ext_vector_type(8)))  __bf16 v8bf;
typedef __attribute__((ext_vector_type(4)))  __bf16 v4bf;
typedef __attribute__((ext_vector_type(8)))  float  v8f;
typedef __attribute__((ext_vector_type(4)))  float  v4f;

// Problem constants (fixed by reference setup_inputs)
#define BB 8
#define CC 256
#define CK 32       // key/value channels
#define HH 128
#define WW 128
#define HW (HH*WW)  // 16384

// async global->LDS copy (CDNA5, ASYNCcnt); inline asm is portable across
// ROCm-7.2 / amdgpu-toolchain builtin arity differences.
__device__ __forceinline__ void async_ld_b128(uint32_t lds_off, const float* gp) {
    asm volatile("global_load_async_to_lds_b128 %0, %1, off"
                 :: "v"(lds_off), "v"(gp) : "memory");
}
__device__ __forceinline__ void wait_asynccnt0() {
    asm volatile("s_wait_asynccnt 0x0" ::: "memory");
}

// ============================================================================
// Kernel 1: k1 = ReLU(x·Wk1^T + bk1), v = x·Wv^T + bv   (bf16 WMMA GEMM)
// One block = one image row (128 pixels). M=128 px, N=64 out ch, K=256.
// ============================================================================
__global__ __launch_bounds__(256) void kv_gemm_kernel(
    const float* __restrict__ x, const float* __restrict__ Wk1, const float* __restrict__ bk1,
    const float* __restrict__ Wv, const float* __restrict__ bv,
    float* __restrict__ k1, float* __restrict__ v)
{
    __shared__ __bf16 xs[128 * 256];   // [pixel][channel], 64 KB
    __shared__ __bf16 wl[64 * 256];    // [outch][channel], 32 KB (Wk1 | Wv)
    __shared__ float  bl[64];

    const int t   = threadIdx.x;
    const int blk = blockIdx.x;          // 0..1023 = B*H
    const int b   = blk >> 7;
    const int h   = blk & 127;
    const float* xb = x + ((size_t)b * CC) * HW + (size_t)h * WW;

    __builtin_prefetch(xb, 0, 0);        // global_prefetch_b8

    // stage x row-tile: 128 px x 256 ch, transpose to [p][c]; process channel
    // pairs so each transposed store is a packed bf16x2 (b32) instead of 2xb16
    for (int it = 0; it < 16; ++it) {
        int f  = it * 256 + t;           // pair-of-float4 id, 4096 total
        int c2 = f >> 5;                 // channel pair 0..127
        int p4 = (f & 31) << 2;
        v4f a0 = *(const v4f*)(xb + (size_t)(2 * c2 + 0) * HW + p4);
        v4f a1 = *(const v4f*)(xb + (size_t)(2 * c2 + 1) * HW + p4);
        #pragma unroll
        for (int i = 0; i < 4; ++i) {
            union { __bf16 h2[2]; uint32_t u; } pk;
            pk.h2[0] = (__bf16)a0[i];
            pk.h2[1] = (__bf16)a1[i];
            *(uint32_t*)&xs[(p4 + i) * 256 + 2 * c2] = pk.u;
        }
    }
    // stage weights: o<32 -> Wk1, o>=32 -> Wv
    for (int it = 0; it < 16; ++it) {
        int f  = it * 256 + t;           // float4 id, 4096 total
        int o  = f >> 6;
        int c4 = (f & 63) << 2;
        const float* wp = (o < 32) ? (Wk1 + o * 256 + c4) : (Wv + (o - 32) * 256 + c4);
        v4f val = *(const v4f*)wp;
        v4bf b4 = { (__bf16)val.x, (__bf16)val.y, (__bf16)val.z, (__bf16)val.w };
        *(v4bf*)(&wl[o * 256 + c4]) = b4;
    }
    if (t < 64) bl[t] = (t < 32) ? bk1[t] : bv[t - 32];
    __syncthreads();

    const int lane = t & 31;
    const int wave = t >> 5;             // 8 waves -> 8 M-tiles of 16 px
    const int mrow = lane & 15;
    const int hi   = lane >> 4;          // 0/1: K-half split (ISA 7.12.2)
    const int koff = hi * 8;
    const int boff = hi * 16;
    const int p    = wave * 16 + mrow;

    v8f acc[4];
    #pragma unroll
    for (int nt = 0; nt < 4; ++nt)
        #pragma unroll
        for (int r = 0; r < 8; ++r) acc[nt][r] = 0.f;

    #pragma unroll
    for (int kc = 0; kc < 8; ++kc) {     // K = 256 in chunks of 32
        const __bf16* ap = &xs[p * 256 + kc * 32];
        v16bf a = *(const v16bf*)(ap + boff);        // K = boff..boff+15
        #pragma unroll
        for (int nt = 0; nt < 4; ++nt) {
            int o = nt * 16 + mrow;                  // N index per lane
            v16bf bm = *(const v16bf*)(&wl[o * 256 + kc * 32 + boff]);
            acc[nt] = __builtin_amdgcn_wmma_f32_16x16x32_bf16(
                false, a, false, bm, (short)0, acc[nt], false, false);
        }
    }

    // D layout: VGPR r -> M = r + hi*8, N = lane&15 ; M contiguous in memory
    #pragma unroll
    for (int nt = 0; nt < 4; ++nt) {
        int o = nt * 16 + mrow;
        float bias = bl[o];
        float* dst = (o < 32)
            ? (k1 + ((size_t)(b * CK + o)        ) * HW + (size_t)h * WW + wave * 16 + hi * 8)
            : (v  + ((size_t)(b * CK + (o - 32)) ) * HW + (size_t)h * WW + wave * 16 + hi * 8);
        float vals[8];
        #pragma unroll
        for (int r = 0; r < 8; ++r) {
            float f = acc[nt][r] + bias;
            if (o < 32) f = fmaxf(f, 0.f);           // ReLU only on key branch
            vals[r] = f;
        }
        *(v4f*)(dst + 0) = (v4f){vals[0], vals[1], vals[2], vals[3]};
        *(v4f*)(dst + 4) = (v4f){vals[4], vals[5], vals[6], vals[7]};
    }
}

// ============================================================================
// Kernel 2: depthwise 3x3 + 1x1 (32->9) + softmax over 9.
// Block = 2 output rows; halo tile k1[32][4][128] staged into LDS via
// GLOBAL_LOAD_ASYNC_TO_LDS_B128 (ASYNCcnt). 9-tap conv runs out of LDS.
// ============================================================================
__global__ __launch_bounds__(256) void attn_kernel(
    const float* __restrict__ k1, const float* __restrict__ Wdw, const float* __restrict__ bdw,
    const float* __restrict__ Wk3, const float* __restrict__ bk3, float* __restrict__ wts)
{
    __shared__ float ks[CK][4][WW];      // 64 KB halo tile: rows h0-1 .. h0+2
    __shared__ float wdwl[32 * 9];
    __shared__ float bdwl[32];
    __shared__ float wk3l[9 * 32];
    __shared__ float bk3l[9];

    const int t   = threadIdx.x;
    const int blk = blockIdx.x;                  // 0..511 = B*H/2
    const int b   = blk >> 6;
    const int h0  = (blk & 63) << 1;

    if (t < 288) { wdwl[t] = Wdw[t]; wk3l[t] = Wk3[t]; }
    if (t < 32)  bdwl[t] = bdw[t];
    if (t < 9)   bk3l[t] = bk3[t];

    // stage halo tile: 32 ch x 4 rows x 128 px (zero-fill out-of-image rows)
    const float* kb = k1 + ((size_t)b * CK) * HW;
    for (int it = 0; it < 16; ++it) {
        int f  = it * 256 + t;                   // float4 id, 4096 total
        int c  = f >> 7;                         // 128 float4s per channel
        int r  = (f >> 5) & 3;                   // halo row 0..3
        int p4 = (f & 31) << 2;
        int hh = h0 - 1 + r;                     // wave-uniform validity
        if ((unsigned)hh < (unsigned)HH) {
            uint32_t lds_off = (uint32_t)(uintptr_t)&ks[c][r][p4];
            async_ld_b128(lds_off, kb + (size_t)c * HW + hh * WW + p4);
        } else {
            *(v4f*)&ks[c][r][p4] = (v4f){0.f, 0.f, 0.f, 0.f};
        }
    }
    wait_asynccnt0();
    __syncthreads();

    const int h  = h0 + (t >> 7);
    const int lr = 1 + (t >> 7);                 // local row in halo tile
    const int w  = t & 127;

    float logit[9];
    #pragma unroll
    for (int j = 0; j < 9; ++j) logit[j] = bk3l[j];

    for (int c = 0; c < CK; ++c) {
        float d = bdwl[c];
        #pragma unroll
        for (int dy = -1; dy <= 1; ++dy) {
            const float* row = &ks[c][lr + dy][0];
            #pragma unroll
            for (int dx = -1; dx <= 1; ++dx) {
                int ww = w + dx;
                if ((unsigned)ww < (unsigned)WW)
                    d += wdwl[c * 9 + (dy + 1) * 3 + (dx + 1)] * row[ww];
            }
        }
        #pragma unroll
        for (int j = 0; j < 9; ++j) logit[j] += wk3l[j * 32 + c] * d;
    }
    // softmax over 9
    float mx = logit[0];
    #pragma unroll
    for (int j = 1; j < 9; ++j) mx = fmaxf(mx, logit[j]);
    float e[9], sum = 0.f;
    #pragma unroll
    for (int j = 0; j < 9; ++j) { e[j] = __expf(logit[j] - mx); sum += e[j]; }
    float inv = 1.f / sum;
    size_t base = ((size_t)b * 9) * HW + (size_t)h * WW + w;
    #pragma unroll
    for (int j = 0; j < 9; ++j) wts[base + (size_t)j * HW] = e[j] * inv;
}

// ============================================================================
// Kernel 3: y = sum_j wts[j]*v[neighbor_j];  out = x + y·Wo^T + bo (WMMA, K=32)
// One block = one image row. M=128 px, N=256, K=32 (single WMMA k-step).
// ============================================================================
__global__ __launch_bounds__(256) void out_gemm_kernel(
    const float* __restrict__ x, const float* __restrict__ v, const float* __restrict__ wts,
    const float* __restrict__ Wo, const float* __restrict__ bo, float* __restrict__ out)
{
    __shared__ __bf16 ys[128 * 32];      // [pixel][cv], 8 KB
    __shared__ __bf16 wol[256 * 32];     // [co][cv],   16 KB
    __shared__ float  bol[256];

    const int t   = threadIdx.x;
    const int blk = blockIdx.x;          // 0..1023 = B*H
    const int b   = blk >> 7;
    const int h   = blk & 127;

    // stage Wo as bf16 [co][cv]
    for (int it = 0; it < 8; ++it) {
        int f  = it * 256 + t;           // 2048 float4s
        int o  = f >> 3;
        int c4 = (f & 7) << 2;
        v4f val = *(const v4f*)(Wo + o * 32 + c4);
        v4bf b4 = { (__bf16)val.x, (__bf16)val.y, (__bf16)val.z, (__bf16)val.w };
        *(v4bf*)(&wol[o * 32 + c4]) = b4;
    }
    bol[t] = bo[t];

    // neighbor-weighted aggregation of v into ys (bf16)
    const float* vb = v   + ((size_t)b * CK) * HW;
    const float* wb = wts + ((size_t)b * 9)  * HW + (size_t)h * WW;
    for (int it = 0; it < 16; ++it) {
        int f  = it * 256 + t;           // 4096 = 128 px * 32 cv
        int p  = f & 127;
        int cv = (f >> 7) & 31;
        float acc = 0.f;
        #pragma unroll
        for (int j = 0; j < 9; ++j) {
            int dy = j / 3 - 1, dx = j % 3 - 1;
            int hh = h + dy, ww = p + dx;
            if ((unsigned)hh < (unsigned)HH && (unsigned)ww < (unsigned)WW)
                acc += wb[(size_t)j * HW + p] * vb[(size_t)cv * HW + hh * WW + ww];
        }
        ys[p * 32 + cv] = (__bf16)acc;
    }
    __syncthreads();

    const int lane = t & 31;
    const int wave = t >> 5;
    const int mrow = lane & 15;
    const int hi   = lane >> 4;
    const int koff = hi * 8;
    const int boff = hi * 16;
    const int p    = wave * 16 + mrow;

    // A fragment (16 px x 32 cv), single K chunk
    v16bf a = *(const v16bf*)(&ys[p * 32 + boff]);

    const float* xb = x   + ((size_t)b * CC) * HW + (size_t)h * WW;
    float*       ob = out + ((size_t)b * CC) * HW + (size_t)h * WW;

    #pragma unroll 4
    for (int nt = 0; nt < 16; ++nt) {    // N = 256 in 16 tiles
        int o = nt * 16 + mrow;
        v16bf bm = *(const v16bf*)(&wol[o * 32 + boff]);
        v8f acc;
        #pragma unroll
        for (int r = 0; r < 8; ++r) acc[r] = 0.f;
        acc = __builtin_amdgcn_wmma_f32_16x16x32_bf16(
            false, a, false, bm, (short)0, acc, false, false);

        float bias = bol[o];
        const float* xr = xb + (size_t)o * HW + wave * 16 + hi * 8;
        float*       od = ob + (size_t)o * HW + wave * 16 + hi * 8;
        v4f x0 = *(const v4f*)(xr);
        v4f x1 = *(const v4f*)(xr + 4);
        v4f r0 = { acc[0] + bias + x0.x, acc[1] + bias + x0.y,
                   acc[2] + bias + x0.z, acc[3] + bias + x0.w };
        v4f r1 = { acc[4] + bias + x1.x, acc[5] + bias + x1.y,
                   acc[6] + bias + x1.z, acc[7] + bias + x1.w };
        *(v4f*)od       = r0;
        *(v4f*)(od + 4) = r1;
    }
}

// ============================================================================
extern "C" void kernel_launch(void* const* d_in, const int* in_sizes, int n_in,
                              void* d_out, int out_size, void* d_ws, size_t ws_size,
                              hipStream_t stream) {
    const float* x   = (const float*)d_in[0];
    const float* Wk1 = (const float*)d_in[1];
    const float* bk1 = (const float*)d_in[2];
    const float* Wdw = (const float*)d_in[3];
    const float* bdw = (const float*)d_in[4];
    const float* Wk3 = (const float*)d_in[5];
    const float* bk3 = (const float*)d_in[6];
    const float* Wv  = (const float*)d_in[7];
    const float* bv  = (const float*)d_in[8];
    const float* Wo  = (const float*)d_in[9];
    const float* bo  = (const float*)d_in[10];
    float* out = (float*)d_out;

    // workspace: k1 (16 MiB) | v (16 MiB) | wts (4.5 MiB) -- all L2-resident
    float* k1w = (float*)d_ws;
    float* vw  = k1w + (size_t)BB * CK * HW;
    float* ww  = vw  + (size_t)BB * CK * HW;

    kv_gemm_kernel<<<BB * HH, 256, 0, stream>>>(x, Wk1, bk1, Wv, bv, k1w, vw);
    attn_kernel<<<BB * HH / 2, 256, 0, stream>>>(k1w, Wdw, bdw, Wk3, bk3, ww);
    out_gemm_kernel<<<BB * HH, 256, 0, stream>>>(x, vw, ww, Wo, bo, out);
}
